// MetaCNNLSTM_EncoderMOE_42786464203495
// MI455X (gfx1250) — matrix operand order repaired and checked by reading, not compiled, not run on hardware
//
#include <hip/hip_runtime.h>
#include <hip/hip_bf16.h>

// ---------------- problem constants ----------------
#define B_    256
#define C_    16
#define T_    1024
#define CC_   64
#define HID_  64
#define RDIM_ 32
#define E_    8
#define H_    128
#define NCLS_ 10
#define K_    5
#define TSTRIP_ 128       // t-strip per conv block
#define PSTR_ 104         // im2col patch row stride (halves): 96 used + 8 pad (208B = 52 banks)

typedef __attribute__((ext_vector_type(16))) _Float16 v16h;
typedef __attribute__((ext_vector_type(8)))  _Float16 v8h;
typedef __attribute__((ext_vector_type(8)))  float    v8f;

static __device__ __forceinline__ v8f wmma16(v16h a, v16h b, v8f c) {
  return __builtin_amdgcn_wmma_f32_16x16x32_f16(
      false, a, false, b, (short)0, c, false, false);
}
static __device__ __forceinline__ v16h join8(v8h lo, v8h hi) {
  return __builtin_shufflevector(lo, hi, 0,1,2,3,4,5,6,7,8,9,10,11,12,13,14,15);
}
static __device__ __forceinline__ float gelu_f(float x) {
  float x3 = x * x * x;
  return 0.5f * x * (1.f + tanhf(0.7978845608f * (x + 0.044715f * x3)));
}
static __device__ __forceinline__ float sigmoid_f(float x) {
  return 1.f / (1.f + __expf(-x));
}

// ---------------- workspace layout (bytes, 256-aligned) ----------------
#define OFF_XH     ((size_t)0)                        // x f16:      B*C*T*2      = 8 MB
#define OFF_HCONV  ((size_t)8388608)                  // hconv f32:  B*CC*T*4     = 64 MB
#define OFF_FEATS  ((size_t)(8388608+67108864))       // feats f16:  T*B*CC*2     = 32 MB
#define OFF_R      ((size_t)(OFF_FEATS+33554432))     // r f32:      B*CC*4
#define OFF_HARD   ((size_t)(OFF_R+65536))            // hard f32:   B*E*4
#define OFF_HT     ((size_t)(OFF_HARD+8192))          // hT f32:     B*H*4
#define OFF_WBCONV ((size_t)(OFF_HT+131072))          // 10 convs * 3*4*32*16 f16 = 120 KB
#define OFF_WLSTM  ((size_t)(OFF_WBCONV+122880))      // 6*32*32*16 f16 = 192 KB

// ================= prep kernels =================
__global__ void prep_x_kernel(const float* __restrict__ x, _Float16* __restrict__ xh, int n) {
  int i = blockIdx.x * blockDim.x + threadIdx.x;
  if (i < n) xh[i] = (_Float16)x[i];
}

// conv weights -> WMMA B-fragment order. conv 0 = router, 1..8 = experts, 9 = shared.
__global__ void prep_convw_kernel(const float* __restrict__ Wc_r,
                                  const float* __restrict__ We,
                                  const float* __restrict__ Ws,
                                  _Float16* __restrict__ wB) {
  int idx = blockIdx.x * blockDim.x + threadIdx.x;
  if (idx >= 10 * 6144) return;
  int conv = idx / 6144, rem = idx % 6144;
  int kb = rem >> 11;            // 0..2
  int nt = (rem >> 9) & 3;       // 0..3
  int lane = (rem >> 4) & 31;    // 0..31
  int j = rem & 15;              // 0..15
  int kl = (j < 8) ? 8 * (lane >> 4) + j : 16 + 8 * (lane >> 4) + (j - 8); // B-frag: k = 8*g + j (j<8), 16 + 8*g + j-8 (j>=8)
  int kk = kb * 32 + kl;
  int o = nt * 16 + (lane & 15);
  float v = 0.f;
  if (kk < C_ * K_) {
    int i = kk / K_, dk = kk % K_;
    const float* w = (conv == 0) ? Wc_r
                   : (conv <= 8) ? (We + (size_t)(conv - 1) * CC_ * C_ * K_)
                                 : Ws;
    v = w[(o * C_ + i) * K_ + dk];
  }
  wB[(size_t)conv * 6144 + (((kb * 4 + nt) * 32 + lane) << 4) + j] = (_Float16)v;
}

// [Wih ; Whh] (K=192 = 6x32) -> WMMA B-fragment order
__global__ void prep_lstmw_kernel(const float* __restrict__ Wih,
                                  const float* __restrict__ Whh,
                                  _Float16* __restrict__ wl) {
  int idx = blockIdx.x * blockDim.x + threadIdx.x;
  if (idx >= 6 * 32 * 32 * 16) return;
  int kb = idx >> 14;            // 0..5
  int nt = (idx >> 9) & 31;      // 0..31
  int lane = (idx >> 4) & 31;
  int j = idx & 15;
  int kl = (j < 8) ? 8 * (lane >> 4) + j : 16 + 8 * (lane >> 4) + (j - 8);
  int kk = kb * 32 + kl;         // 0..191
  int go = nt * 16 + (lane & 15);
  float v = (kk < CC_) ? Wih[go * CC_ + kk] : Whh[go * H_ + (kk - CC_)];
  wl[(((size_t)(kb * 32 + nt) * 32 + lane) << 4) + j] = (_Float16)v;
}

// ---- shared helper: stage x strip + build im2col patch in LDS ----
static __device__ __forceinline__ void build_patch(
    const _Float16* __restrict__ xh, int b, int ts,
    _Float16* __restrict__ xs, _Float16* __restrict__ patch, int tid) {
  for (int i = tid; i < C_ * (TSTRIP_ + 4); i += 256) {
    int c = i / (TSTRIP_ + 4), tl = i % (TSTRIP_ + 4);
    int gt = ts + tl - 2;
    xs[i] = (gt >= 0 && gt < T_) ? xh[((size_t)b * C_ + c) * T_ + gt] : (_Float16)0.f;
  }
  __syncthreads();
  int t = tid >> 1, half = tid & 1;
  _Float16* prow = patch + t * PSTR_;
  const _Float16* xrow = xs + t;
#pragma unroll
  for (int ii = 0; ii < 8; ++ii) {
    int i = half * 8 + ii;
#pragma unroll
    for (int dk = 0; dk < K_; ++dk)
      prow[i * K_ + dk] = xrow[i * (TSTRIP_ + 4) + dk];
  }
#pragma unroll
  for (int q = 0; q < 8; ++q)
    prow[80 + half * 8 + q] = (_Float16)0.f;
  __syncthreads();
}

static __device__ __forceinline__ v16h patch_afrag(const _Float16* prow, int kb, int g) {
  v8h lo = *(const v8h*)(prow + kb * 32 + 8 * g);
  v8h hi = *(const v8h*)(prow + kb * 32 + 16 + 8 * g);
  return join8(lo, hi);
}

// ================= router conv (WMMA implicit GEMM) =================
__global__ __launch_bounds__(256) void router_conv_kernel(
    const _Float16* __restrict__ xh, const _Float16* __restrict__ wB,
    float* __restrict__ hconv) {
  const int blk = blockIdx.x;
  const int b = blk / (T_ / TSTRIP_);
  const int ts = (blk % (T_ / TSTRIP_)) * TSTRIP_;
  const int tid = threadIdx.x, wave = tid >> 5, lane = tid & 31;
  const int m = lane & 15, g = lane >> 4;

  __shared__ __align__(16) _Float16 xs[C_ * (TSTRIP_ + 4)];
  __shared__ __align__(16) _Float16 patch[TSTRIP_ * PSTR_];
  build_patch(xh, b, ts, xs, patch, tid);

  const _Float16* prow = patch + (wave * 16 + m) * PSTR_;
  v16h af[3];
#pragma unroll
  for (int kb = 0; kb < 3; ++kb) af[kb] = patch_afrag(prow, kb, g);

#pragma unroll
  for (int nt = 0; nt < 4; ++nt) {
    v8f acc = {};
#pragma unroll
    for (int kb = 0; kb < 3; ++kb) {
      v16h bf = *(const v16h*)(wB + (((kb * 4 + nt) * 32 + lane) << 4));
      acc = wmma16(af[kb], bf, acc);
    }
    int o = nt * 16 + (lane & 15);
    int t0 = ts + wave * 16 + 8 * g;
#pragma unroll
    for (int r = 0; r < 8; ++r)
      hconv[((size_t)b * CC_ + o) * T_ + (t0 + r)] = acc[r];
  }
}

// ================= GroupNorm -> GELU -> GAP =================
__global__ __launch_bounds__(256) void gn_gap_kernel(
    const float* __restrict__ hconv, const float* __restrict__ gng,
    const float* __restrict__ gnb, float* __restrict__ rout) {
  int b = blockIdx.x, tid = threadIdx.x;
  __shared__ float s_a[256], s_b[256];
  __shared__ float s_m[8], s_rs[8];
  int gp = tid >> 5, sl = tid & 31;
  const float* base = hconv + ((size_t)b * CC_ + gp * 8) * T_;
  float sm = 0.f, sq = 0.f;
  for (int i = sl; i < 8 * T_; i += 32) { float v = base[i]; sm += v; sq += v * v; }
  s_a[tid] = sm; s_b[tid] = sq;
  __syncthreads();
  if (sl == 0) {
    float a = 0.f, c = 0.f;
    for (int k = 0; k < 32; ++k) { a += s_a[tid + k]; c += s_b[tid + k]; }
    float mean = a / (8.f * T_), var = c / (8.f * T_) - mean * mean;
    s_m[gp] = mean; s_rs[gp] = rsqrtf(var + 1e-5f);
  }
  __syncthreads();
  int c = tid >> 2, q = tid & 3;
  float mm = s_m[c >> 3], rs = s_rs[c >> 3], gg = gng[c], bb = gnb[c];
  const float* cb = hconv + ((size_t)b * CC_ + c) * T_;
  float acc = 0.f;
  for (int t = q * (T_ / 4); t < (q + 1) * (T_ / 4); ++t)
    acc += gelu_f((cb[t] - mm) * rs * gg + bb);
  s_a[tid] = acc;
  __syncthreads();
  if (q == 0)
    rout[b * CC_ + c] = (s_a[tid] + s_a[tid + 1] + s_a[tid + 2] + s_a[tid + 3]) * (1.f / T_);
}

// ================= router MLP + gate + top-2 =================
__global__ __launch_bounds__(64) void router_mlp_kernel(
    const float* __restrict__ rin, const float* __restrict__ Wm1, const float* __restrict__ bm1,
    const float* __restrict__ lng, const float* __restrict__ lnb,
    const float* __restrict__ Wm2, const float* __restrict__ bm2,
    const float* __restrict__ Wg, const float* __restrict__ bg,
    float* __restrict__ hard) {
  int b = blockIdx.x, j = threadIdx.x;
  __shared__ float rv[64], h1[64], r2[32], st[2];
  rv[j] = rin[b * 64 + j];
  __syncthreads();
  float acc = bm1[j];
  for (int k = 0; k < 64; ++k) acc += rv[k] * Wm1[k * 64 + j];
  h1[j] = acc;
  __syncthreads();
  if (j == 0) {
    float s = 0.f, ss = 0.f;
    for (int k = 0; k < 64; ++k) { s += h1[k]; ss += h1[k] * h1[k]; }
    float mean = s / 64.f;
    st[0] = mean; st[1] = rsqrtf(ss / 64.f - mean * mean + 1e-5f);
  }
  __syncthreads();
  float hn = gelu_f((h1[j] - st[0]) * st[1] * lng[j] + lnb[j]);
  __syncthreads();
  h1[j] = hn;
  __syncthreads();
  if (j < 32) {
    float a = bm2[j];
    for (int k = 0; k < 64; ++k) a += h1[k] * Wm2[k * 32 + j];
    r2[j] = a;
  }
  __syncthreads();
  if (j == 0) {
    float p[8];
    float mx = -1e30f;
    for (int e = 0; e < 8; ++e) {
      float a = bg[e];
      for (int k = 0; k < 32; ++k) a += r2[k] * Wg[k * 8 + e];
      p[e] = a; mx = fmaxf(mx, a);
    }
    float s = 0.f;
    for (int e = 0; e < 8; ++e) { p[e] = __expf(p[e] - mx); s += p[e]; }
    for (int e = 0; e < 8; ++e) p[e] /= s;
    int i1 = 0;
    for (int e = 1; e < 8; ++e) if (p[e] > p[i1]) i1 = e;
    int i2 = (i1 == 0) ? 1 : 0;
    for (int e = 0; e < 8; ++e) if (e != i1 && p[e] > p[i2]) i2 = e;
    float hs = p[i1] + p[i2] + 1e-9f;
    for (int e = 0; e < 8; ++e)
      hard[b * 8 + e] = (e == i1 || e == i2) ? p[e] / hs : 0.f;
  }
}

// ================= expert + shared convs, gated mix (WMMA) =================
__global__ __launch_bounds__(256) void expert_conv_kernel(
    const _Float16* __restrict__ xh, const _Float16* __restrict__ wB,
    const float* __restrict__ be, const float* __restrict__ bs,
    const float* __restrict__ hard, _Float16* __restrict__ feats_tm) {
  const int blk = blockIdx.x;
  const int b = blk / (T_ / TSTRIP_);
  const int ts = (blk % (T_ / TSTRIP_)) * TSTRIP_;
  const int tid = threadIdx.x, wave = tid >> 5, lane = tid & 31;
  const int m = lane & 15, g = lane >> 4;

  __shared__ __align__(16) _Float16 xs[C_ * (TSTRIP_ + 4)];
  __shared__ __align__(16) _Float16 patch[TSTRIP_ * PSTR_];
  build_patch(xh, b, ts, xs, patch, tid);

  float gw[E_];
#pragma unroll
  for (int e = 0; e < E_; ++e) gw[e] = hard[b * E_ + e];

  const _Float16* prow = patch + (wave * 16 + m) * PSTR_;
  v16h af[3];
#pragma unroll
  for (int kb = 0; kb < 3; ++kb) af[kb] = patch_afrag(prow, kb, g);

  float mix[4][8];
#pragma unroll
  for (int nt = 0; nt < 4; ++nt)
#pragma unroll
    for (int r = 0; r < 8; ++r) mix[nt][r] = 0.f;

  for (int e = 0; e < 9; ++e) {
    float w; const _Float16* wBe; const float* bias;
    if (e < E_) {
      w = gw[e];
      if (w == 0.f) continue;            // top-2 sparsity: skip unselected experts
      wBe = wB + (size_t)(1 + e) * 6144;
      bias = be + e * CC_;
    } else {
      w = 1.f; wBe = wB + (size_t)9 * 6144; bias = bs;
    }
#pragma unroll
    for (int nt = 0; nt < 4; ++nt) {
      v8f acc = {};
#pragma unroll
      for (int kb = 0; kb < 3; ++kb) {
        v16h bf = *(const v16h*)(wBe + (((kb * 4 + nt) * 32 + lane) << 4));
        acc = wmma16(af[kb], bf, acc);
      }
      float bv = bias[nt * 16 + (lane & 15)];
#pragma unroll
      for (int r = 0; r < 8; ++r) mix[nt][r] += w * gelu_f(acc[r] + bv);
    }
  }

#pragma unroll
  for (int nt = 0; nt < 4; ++nt) {
    int o = nt * 16 + (lane & 15);
    int t0 = ts + wave * 16 + 8 * g;
#pragma unroll
    for (int r = 0; r < 8; ++r)
      feats_tm[((size_t)(t0 + r) * B_ + b) * CC_ + o] = (_Float16)mix[nt][r];
  }
}

// ================= LSTM: 16 independent batch-tile workgroups =================
// Recurrence is elementwise in batch -> no cross-WG coupling. Each block owns a
// 16-row batch tile: h tile (16x128 f16, XOR-chunk-swizzled) in 4 KB LDS, c in
// VGPRs. 8 waves/block (2/SIMD), wave w owns h-columns [16w,16w+16).
// The invariant B matrix (24 fragments = 192 VGPRs/lane) is hoisted into
// registers before the t-loop: the 1024-step critical path has NO weight
// traffic -- only 4 prefetched x loads, 8 ds_load_b128 for h, 24 WMMAs and
// the f32 pointwise update per wave per step.
__global__ __launch_bounds__(256) void lstm_kernel(
    const _Float16* __restrict__ feats_tm, const _Float16* __restrict__ wl,
    const float* __restrict__ bih, const float* __restrict__ bhh,
    float* __restrict__ hT) {
  __shared__ __align__(16) _Float16 hbuf[16 * H_];   // 4 KB
  const int tid = threadIdx.x, wave = tid >> 5, lane = tid & 31;
  const int b0 = blockIdx.x * 16;                    // batch tile base
  const int m = lane & 15, g = lane >> 4, n = lane & 15;

  for (int i = tid; i < 16 * H_; i += 256) hbuf[i] = (_Float16)0.f;

  // hoist invariant B fragments into VGPRs (4 gate quadrants x 6 k-blocks)
  v16h bfr[4][6];
#pragma unroll
  for (int gi = 0; gi < 4; ++gi) {
    int ntile = gi * 8 + wave;
#pragma unroll
    for (int kb = 0; kb < 6; ++kb)
      bfr[gi][kb] = *(const v16h*)(wl + (((size_t)(kb * 32 + ntile) * 32 + lane) << 4));
  }

  float bias[4];
#pragma unroll
  for (int gi = 0; gi < 4; ++gi) {
    int go = (gi * 8 + wave) * 16 + n;
    bias[gi] = bih[go] + bhh[go];
  }
  float cst[8];
#pragma unroll
  for (int r = 0; r < 8; ++r) cst[r] = 0.f;
  __syncthreads();

  const int sw = m & 7;                              // h chunk swizzle (local row m)
  const _Float16* hrow = hbuf + m * H_;

  for (int t = 0; t < T_; ++t) {
    const _Float16* xrow = feats_tm + ((size_t)t * B_ + b0 + m) * CC_;
    if (t + 1 < T_)                                  // global_prefetch_b8 next step's x
      __builtin_prefetch(feats_tm + ((size_t)(t + 1) * B_ + b0 + m) * CC_, 0, 1);

    v16h af[6];
#pragma unroll
    for (int kb = 0; kb < 2; ++kb) {                 // x part: contiguous global 16B loads
      v8h lo = *(const v8h*)(xrow + kb * 32 + 8 * g);
      v8h hi = *(const v8h*)(xrow + kb * 32 + 16 + 8 * g);
      af[kb] = join8(lo, hi);
    }
#pragma unroll
    for (int kb = 2; kb < 6; ++kb) {                 // h part: swizzled 16B ds loads
      int c0 = (2 * kb - 4) ^ sw;
      int c1 = (2 * kb - 3) ^ sw;
      v8h lo = *(const v8h*)(hrow + c0 * 16 + 8 * g);
      v8h hi = *(const v8h*)(hrow + c1 * 16 + 8 * g);
      af[kb] = join8(lo, hi);
    }
    __syncthreads();                                 // all h reads done before writes

    v8f acc[4];
#pragma unroll
    for (int gi = 0; gi < 4; ++gi) {
      v8f a2 = {};
#pragma unroll
      for (int kb = 0; kb < 6; ++kb)
        a2 = wmma16(af[kb], bfr[gi][kb], a2);        // weights from registers
      acc[gi] = a2;
    }
#pragma unroll
    for (int r = 0; r < 8; ++r) {
      float iv = sigmoid_f(acc[0][r] + bias[0]);
      float fv = sigmoid_f(acc[1][r] + bias[1]);
      float gv = tanhf    (acc[2][r] + bias[2]);
      float ov = sigmoid_f(acc[3][r] + bias[3]);
      float c = fv * cst[r] + iv * gv;               // c resident in VGPRs
      cst[r] = c;
      int bb = 8 * g + r;                            // local batch row
      hbuf[bb * H_ + ((wave ^ (bb & 7)) * 16) + n] = (_Float16)(ov * tanhf(c));
    }
    __syncthreads();                                 // h writes visible before next step
  }
  // de-swizzle final h -> hT (f32)
  for (int i = tid; i < 16 * H_; i += 256) {
    int lr = i >> 7, k = i & 127;
    hT[(b0 + lr) * H_ + k] =
        (float)hbuf[lr * H_ + (((k >> 4) ^ (lr & 7)) << 4) + (k & 15)];
  }
}

// ================= classifier head =================
__global__ void head_kernel(const float* __restrict__ hT, const float* __restrict__ Wh,
                            const float* __restrict__ bh, float* __restrict__ out) {
  int idx = blockIdx.x * blockDim.x + threadIdx.x;
  if (idx >= B_ * NCLS_) return;
  int b = idx / NCLS_, n = idx % NCLS_;
  float a = bh[n];
  for (int k = 0; k < H_; ++k) a += hT[b * H_ + k] * Wh[k * NCLS_ + n];
  out[idx] = a;
}

// ================= launch =================
extern "C" void kernel_launch(void* const* d_in, const int* in_sizes, int n_in,
                              void* d_out, int out_size, void* d_ws, size_t ws_size,
                              hipStream_t stream) {
  const float* x    = (const float*)d_in[0];
  const float* Wc_r = (const float*)d_in[1];
  const float* gng  = (const float*)d_in[2];
  const float* gnb  = (const float*)d_in[3];
  const float* Wm1  = (const float*)d_in[4];
  const float* bm1  = (const float*)d_in[5];
  const float* lng  = (const float*)d_in[6];
  const float* lnb  = (const float*)d_in[7];
  const float* Wm2  = (const float*)d_in[8];
  const float* bm2  = (const float*)d_in[9];
  const float* Wg   = (const float*)d_in[10];
  const float* bg   = (const float*)d_in[11];
  const float* We   = (const float*)d_in[12];
  const float* be   = (const float*)d_in[13];
  const float* Ws   = (const float*)d_in[14];
  const float* bs   = (const float*)d_in[15];
  const float* Wih  = (const float*)d_in[16];
  const float* Whh  = (const float*)d_in[17];
  const float* bih  = (const float*)d_in[18];
  const float* bhh  = (const float*)d_in[19];
  const float* Wh   = (const float*)d_in[20];
  const float* bh   = (const float*)d_in[21];
  float* out = (float*)d_out;

  char* ws = (char*)d_ws;
  _Float16* xh    = (_Float16*)(ws + OFF_XH);
  float*    hconv = (float*)   (ws + OFF_HCONV);
  _Float16* feats = (_Float16*)(ws + OFF_FEATS);
  float*    rbuf  = (float*)   (ws + OFF_R);
  float*    hard  = (float*)   (ws + OFF_HARD);
  float*    hTbuf = (float*)   (ws + OFF_HT);
  _Float16* wB    = (_Float16*)(ws + OFF_WBCONV);
  _Float16* wl    = (_Float16*)(ws + OFF_WLSTM);

  {
    int n = B_ * C_ * T_;
    prep_x_kernel<<<(n + 255) / 256, 256, 0, stream>>>(x, xh, n);
  }
  prep_convw_kernel<<<(10 * 6144 + 255) / 256, 256, 0, stream>>>(Wc_r, We, Ws, wB);
  prep_lstmw_kernel<<<(6 * 32 * 32 * 16 + 255) / 256, 256, 0, stream>>>(Wih, Whh, wl);

  router_conv_kernel<<<B_ * (T_ / TSTRIP_), 256, 0, stream>>>(xh, wB, hconv);
  gn_gap_kernel<<<B_, 256, 0, stream>>>(hconv, gng, gnb, rbuf);
  router_mlp_kernel<<<B_, 64, 0, stream>>>(rbuf, Wm1, bm1, lng, lnb, Wm2, bm2, Wg, bg, hard);

  expert_conv_kernel<<<B_ * (T_ / TSTRIP_), 256, 0, stream>>>(xh, wB, be, bs, hard, feats);

  // 16 independent batch tiles x 8 waves; recurrence is batch-parallel
  lstm_kernel<<<16, 256, 0, stream>>>(feats, wl, bih, bhh, hTbuf);

  head_kernel<<<(B_ * NCLS_ + 255) / 256, 256, 0, stream>>>(hTbuf, Wh, bh, out);
}
